// RINModel_75892072120997
// MI455X (gfx1250) — matrix operand order repaired and checked by reading, not compiled, not run on hardware
//
#include <hip/hip_runtime.h>
#include <math.h>

// ---------------- problem sizes ----------------
#define V_SZ   50257
#define D_SZ   128
#define L_SZ   2
#define N_SZ   256
#define RES_SZ 4096
#define B_SZ   2
#define S_SZ   512
#define BS_SZ  (B_SZ * S_SZ)          // 1024 tokens
#define NT_V   ((V_SZ + 15) / 16)     // 3142 vocab tiles
#define MT_TOK (BS_SZ / 16)           // 64 token tiles
#define KT_D   (D_SZ / 32)            // 4 K-tiles of 32
#define TILE_DW 256                   // 32 lanes * 8 dwords per wmma operand tile

#define PHI_F 1.6180339887498949f
#define K2P_F 651.8986469044033f      // RES / (2*pi)
#define G2P_F 0.0015339807878856412f  // 2*pi / RES

// CDNA5 WMMA vector types (wave32; 16x16x32 bf16 -> f32)
typedef __attribute__((ext_vector_type(16))) __bf16        v16bf;
typedef __attribute__((ext_vector_type(8)))  float         v8f;
typedef __attribute__((ext_vector_type(8)))  unsigned int  v8u;

#define WMMA_BF16(A, B, C) \
    __builtin_amdgcn_wmma_f32_16x16x32_bf16(false, (A), false, (B), (short)0, (C), false, false)

__device__ __forceinline__ v16bf as_bf(v8u u) {
    union { v8u u; v16bf b; } c; c.u = u; return c.b;
}

// ---------------- workspace layout (bytes) ----------------
static constexpr size_t OFF_TAB = 0;                                              // RES float2
static constexpr size_t OFF_XR  = OFF_TAB + (size_t)RES_SZ * 8;                   // BS*D f32
static constexpr size_t OFF_XI  = OFF_XR  + (size_t)BS_SZ * D_SZ * 4;
static constexpr size_t OFF_XC  = OFF_XI  + (size_t)BS_SZ * D_SZ * 4;
static constexpr size_t OFF_COS = OFF_XC  + (size_t)BS_SZ * D_SZ * 4;             // BS*N f32
static constexpr size_t OFF_SIN = OFF_COS + (size_t)BS_SZ * N_SZ * 4;
static constexpr size_t OFF_PK  = OFF_SIN + (size_t)BS_SZ * N_SZ * 4;             // L*N*D float4
static constexpr size_t OFF_AR  = OFF_PK  + (size_t)L_SZ * N_SZ * D_SZ * 16;      // A tiles (xr)
static constexpr size_t OFF_AI  = OFF_AR  + (size_t)MT_TOK * KT_D * TILE_DW * 4;  // A tiles (xi)
static constexpr size_t OFF_BR  = OFF_AI  + (size_t)MT_TOK * KT_D * TILE_DW * 4;  // B tiles (Wr)
static constexpr size_t OFF_BI  = OFF_BR  + (size_t)NT_V * KT_D * TILE_DW * 4;    // B tiles (Wi)

// float -> bf16 (round-to-nearest-even), packed pair (even k in [15:0])
__device__ __forceinline__ unsigned int pack_bf16(float a, float b) {
    unsigned int ua = __float_as_uint(a);
    unsigned int ub = __float_as_uint(b);
    ua = (ua + 0x7FFFu + ((ua >> 16) & 1u)) >> 16;
    ub = (ub + 0x7FFFu + ((ub >> 16) & 1u)) >> 16;
    return (ua & 0xFFFFu) | (ub << 16);
}

// ---------------- small kernels ----------------
__global__ void build_tab_kernel(float2* tab) {
    int i = blockIdx.x * blockDim.x + threadIdx.x;
    if (i < RES_SZ) {
        float g = (float)i * G2P_F;
        tab[i] = make_float2(__sinf(g), __cosf(g));
    }
}

// Packed coefficients: (a, b, attn_cos, attn_sin) per (l,n,d) -> one b128 load in the hot loop.
// a = (RES/2pi)/(1+|W|), b = Bp * RES/2pi  (pre-scaled so the hot loop is one FMA + floor).
__global__ void coef_kernel(const float* W, const float* Bp, const float* acW,
                            const float* asW, float4* pk) {
    int i = blockIdx.x * blockDim.x + threadIdx.x;
    if (i < L_SZ * N_SZ * D_SZ) {
        pk[i] = make_float4(K2P_F / (1.0f + fabsf(W[i])), Bp[i] * K2P_F, acW[i], asW[i]);
    }
}

__global__ void embed_kernel(const int* ids, const float* emb, float* xr, float* xi) {
    int i = blockIdx.x * blockDim.x + threadIdx.x;
    if (i < BS_SZ * D_SZ) {
        int bs = i >> 7, d = i & (D_SZ - 1);
        long tok = (long)ids[bs];
        xr[i] = emb[tok * (2 * D_SZ) + d];
        xi[i] = emb[tok * (2 * D_SZ) + D_SZ + d];
    }
}

// xc = concat(xr,xi) @ cw[l]^T + cb[l]    (1024 x 256) x (256 x 128), LDS-staged
__global__ __launch_bounds__(128) void contract_kernel(const float* xr, const float* xi,
                                                       const float* cw, const float* cb,
                                                       float* xc, int l) {
    __shared__ float vec[2 * D_SZ];
    int bs = blockIdx.x, t = threadIdx.x;
    vec[t]        = xr[bs * D_SZ + t];
    vec[t + D_SZ] = xi[bs * D_SZ + t];
    __syncthreads();
    const float* row = cw + ((size_t)(l * D_SZ + t)) * (2 * D_SZ);
    float acc = cb[l * D_SZ + t];
    #pragma unroll 8
    for (int k = 0; k < 2 * D_SZ; ++k) acc = fmaf(vec[k], row[k], acc);
    xc[bs * D_SZ + t] = acc;
}

// Heavy phase: per (token, n): reduce over d of LUT-sin/cos(xc*a + b + t) * attn_{sin,cos}
__global__ __launch_bounds__(256) void attn_kernel(const float* xc, const float4* pk,
                                                   const float2* tab,
                                                   float* cosS, float* sinS, int l) {
    __shared__ float2 s_tab[RES_SZ];   // 32 KB fused (sin,cos) LUT -> ds_load_b64
    __shared__ float  s_xc[D_SZ];
    int bs = blockIdx.x, t = threadIdx.x;
    for (int j = t; j < RES_SZ; j += 256) s_tab[j] = tab[j];
    if (t < D_SZ) s_xc[t] = xc[bs * D_SZ + t];
    __syncthreads();

    int n = t;
    float ts = (float)(bs & (S_SZ - 1)) * (PHI_F * K2P_F);
    const float4* row = pk + ((size_t)(l * N_SZ + n)) * D_SZ;
    float cs = 0.0f, ss = 0.0f;
    #pragma unroll 4
    for (int d = 0; d < D_SZ; ++d) {
        float4 c = row[d];                       // one global_load_b128 per element
        float u = fmaf(s_xc[d], c.x, c.y + ts);
        int idx = ((int)floorf(u)) & (RES_SZ - 1);
        float2 sc = s_tab[idx];
        cs = fmaf(sc.y, c.z, cs);
        ss = fmaf(sc.x, c.w, ss);
    }
    cosS[bs * N_SZ + n] = cs;
    sinS[bs * N_SZ + n] = ss;
}

// xr = silu(cosS @ opr[l]^T), xi = silu(sinS @ opi[l]^T)
__global__ __launch_bounds__(128) void outproj_kernel(const float* cosS, const float* sinS,
                                                      const float* opr, const float* opi,
                                                      float* xr, float* xi, int l) {
    __shared__ float cv[N_SZ];
    __shared__ float sv[N_SZ];
    int bs = blockIdx.x, t = threadIdx.x;
    cv[t]       = cosS[bs * N_SZ + t];
    cv[t + 128] = cosS[bs * N_SZ + t + 128];
    sv[t]       = sinS[bs * N_SZ + t];
    sv[t + 128] = sinS[bs * N_SZ + t + 128];
    __syncthreads();
    const float* rr = opr + ((size_t)(l * D_SZ + t)) * N_SZ;
    const float* ri = opi + ((size_t)(l * D_SZ + t)) * N_SZ;
    float ar = 0.0f, ai = 0.0f;
    #pragma unroll 8
    for (int n = 0; n < N_SZ; ++n) {
        ar = fmaf(cv[n], rr[n], ar);
        ai = fmaf(sv[n], ri[n], ai);
    }
    xr[bs * D_SZ + t] = ar / (1.0f + __expf(-ar));
    xi[bs * D_SZ + t] = ai / (1.0f + __expf(-ai));
}

// Repack xr/xi into the ISA 16-bit A-matrix (16x32 MxK) lane/VGPR layout, bf16-packed.
__global__ void prepA_kernel(const float* xr, const float* xi,
                             unsigned int* Ar, unsigned int* Ai) {
    int idx = blockIdx.x * blockDim.x + threadIdx.x;
    if (idx >= MT_TOK * KT_D * TILE_DW) return;
    int v = idx & 7, lane = (idx >> 3) & 31, tk = idx >> 8;
    int kt = tk & 3, mt = tk >> 2;
    int m = mt * 16 + (lane & 15);
    int khalf = lane >> 4;  // lanes 16..31 hold K+8 (lo group) / K+24 (hi group)
    int kb = kt * 32 + ((v >= 4) ? 16 : 0) + khalf * 8 + (v & 3) * 2;
    Ar[idx] = pack_bf16(xr[m * D_SZ + kb], xr[m * D_SZ + kb + 1]);
    Ai[idx] = pack_bf16(xi[m * D_SZ + kb], xi[m * D_SZ + kb + 1]);
}

// Repack Wr/Wi rows into the ISA 16-bit B-matrix (32x16 KxN) layout; pad vocab to 16.
__global__ void prepB_kernel(const float* Wr, const float* Wi,
                             unsigned int* Br, unsigned int* Bi) {
    int idx = blockIdx.x * blockDim.x + threadIdx.x;
    if (idx >= NT_V * KT_D * TILE_DW) return;
    int v = idx & 7, lane = (idx >> 3) & 31, tk = idx >> 8;
    int kt = tk & 3, nt = tk >> 2;
    int n = nt * 16 + (lane & 15);
    int kb = kt * 32 + ((v >= 4) ? 16 : 0) + ((lane >= 16) ? 8 : 0) + (v & 3) * 2;
    float r0 = 0.f, r1 = 0.f, i0 = 0.f, i1 = 0.f;
    if (n < V_SZ) {
        size_t o = (size_t)n * D_SZ + kb;
        r0 = Wr[o]; r1 = Wr[o + 1];
        i0 = Wi[o]; i1 = Wi[o + 1];
    }
    Br[idx] = pack_bf16(r0, r1);
    Bi[idx] = pack_bf16(i0, i1);
}

__device__ __forceinline__ void st_mag(float* out, int m, int n, float lr, float li) {
    __builtin_nontemporal_store(sqrtf(fmaf(lr, lr, fmaf(li, li, 1e-8f))),
                                out + (size_t)m * V_SZ + n);
}

// Vocab projection, 2 m-tiles x 2 n-tiles per wave.
// LDS holds THREE A variants per m-tile: xr, xi, and -xi (sign pre-flipped at staging
// time, 24 KB total). lr = ar*br + (-ai)*bi then accumulates in one WMMA chain with
// no per-iteration VALU on operand registers -> no WMMA->VALU hazard NOPs in the
// hot loop. Output stores are non-temporal (write-once, keep B tiles hot in L2).
__global__ __launch_bounds__(256) void vocab_kernel(const unsigned int* Ar, const unsigned int* Ai,
                                                    const unsigned int* Br, const unsigned int* Bi,
                                                    float* out) {
    __shared__ unsigned int sA[2 * 3 * KT_D * TILE_DW];   // [mi][r|i|negi][kt][lane*8+v], 24 KB
    int wave = threadIdx.x >> 5;
    int lane = threadIdx.x & 31;
    int mp  = blockIdx.y;                  // token-tile pair (2*mp, 2*mp+1)
    int nt0 = blockIdx.x * 16 + wave * 2;  // wave-uniform -> EXEC all-ones for WMMA

    for (int j = threadIdx.x; j < 2 * 2 * KT_D * TILE_DW; j += 256) {
        int e = j & 255, kt = (j >> 8) & 3, ri = (j >> 10) & 1, mi = j >> 11;
        const unsigned int* src = ri ? Ai : Ar;
        unsigned int val = src[((size_t)((2 * mp + mi) * KT_D + kt)) * TILE_DW + e];
        sA[((mi * 3 + ri) * KT_D + kt) * TILE_DW + e] = val;
        if (ri) sA[((mi * 3 + 2) * KT_D + kt) * TILE_DW + e] = val ^ 0x80008000u;
    }
    __syncthreads();
    if (nt0 >= NT_V) return;               // after the only barrier; wave-uniform
    bool n1ok = (nt0 + 1) < NT_V;

    v8f aLr0 = {}, aLr1 = {}, aLr2 = {}, aLr3 = {};  // lr accum: (mi,ni)=(0,0)(0,1)(1,0)(1,1)
    v8f aLi0 = {}, aLi1 = {}, aLi2 = {}, aLi3 = {};  // li accum

    #pragma unroll
    for (int kt = 0; kt < KT_D; ++kt) {
        v16bf ar0 = as_bf(*(const v8u*)&sA[((0 * 3 + 0) * KT_D + kt) * TILE_DW + lane * 8]);
        v16bf ai0 = as_bf(*(const v8u*)&sA[((0 * 3 + 1) * KT_D + kt) * TILE_DW + lane * 8]);
        v16bf an0 = as_bf(*(const v8u*)&sA[((0 * 3 + 2) * KT_D + kt) * TILE_DW + lane * 8]);
        v16bf ar1 = as_bf(*(const v8u*)&sA[((1 * 3 + 0) * KT_D + kt) * TILE_DW + lane * 8]);
        v16bf ai1 = as_bf(*(const v8u*)&sA[((1 * 3 + 1) * KT_D + kt) * TILE_DW + lane * 8]);
        v16bf an1 = as_bf(*(const v8u*)&sA[((1 * 3 + 2) * KT_D + kt) * TILE_DW + lane * 8]);

        size_t b0 = ((size_t)(nt0 * KT_D + kt)) * TILE_DW + lane * 8;
        if (kt + 1 < KT_D) {
            __builtin_prefetch((const void*)(Br + b0 + TILE_DW), 0, 1);
            __builtin_prefetch((const void*)(Bi + b0 + TILE_DW), 0, 1);
        }
        v16bf br0 = *(const v16bf*)(Br + b0);
        v16bf bi0 = *(const v16bf*)(Bi + b0);

        aLr0 = WMMA_BF16(ar0, br0, aLr0);  aLr0 = WMMA_BF16(an0, bi0, aLr0);
        aLi0 = WMMA_BF16(ar0, bi0, aLi0);  aLi0 = WMMA_BF16(ai0, br0, aLi0);
        aLr2 = WMMA_BF16(ar1, br0, aLr2);  aLr2 = WMMA_BF16(an1, bi0, aLr2);
        aLi2 = WMMA_BF16(ar1, bi0, aLi2);  aLi2 = WMMA_BF16(ai1, br0, aLi2);

        if (n1ok) {
            size_t b1 = b0 + (size_t)KT_D * TILE_DW;  // next n-tile, same kt
            v16bf br1 = *(const v16bf*)(Br + b1);
            v16bf bi1 = *(const v16bf*)(Bi + b1);
            aLr1 = WMMA_BF16(ar0, br1, aLr1);  aLr1 = WMMA_BF16(an0, bi1, aLr1);
            aLi1 = WMMA_BF16(ar0, bi1, aLi1);  aLi1 = WMMA_BF16(ai0, br1, aLi1);
            aLr3 = WMMA_BF16(ar1, br1, aLr3);  aLr3 = WMMA_BF16(an1, bi1, aLr3);
            aLi3 = WMMA_BF16(ar1, bi1, aLi3);  aLi3 = WMMA_BF16(ai1, br1, aLi3);
        }
    }

    // C/D layout: VGPR r -> M = r (lanes 0-15) / r+8 (lanes 16-31); N = lane & 15
    int half = lane >> 4, colL = lane & 15;
    int m0 = mp * 32 + half * 8;          // rows of mi=0 tile; mi=1 tile is +16
    int c0 = nt0 * 16 + colL;
    int c1 = c0 + 16;
    #pragma unroll
    for (int r = 0; r < 8; ++r) {
        if (c0 < V_SZ) {
            st_mag(out, m0 + r,      c0, aLr0[r], aLi0[r]);
            st_mag(out, m0 + 16 + r, c0, aLr2[r], aLi2[r]);
        }
        if (n1ok && c1 < V_SZ) {
            st_mag(out, m0 + r,      c1, aLr1[r], aLi1[r]);
            st_mag(out, m0 + 16 + r, c1, aLr3[r], aLi3[r]);
        }
    }
}

// ---------------- launch ----------------
extern "C" void kernel_launch(void* const* d_in, const int* in_sizes, int n_in,
                              void* d_out, int out_size, void* d_ws, size_t ws_size,
                              hipStream_t stream) {
    (void)in_sizes; (void)n_in; (void)out_size; (void)ws_size;
    const int*   ids = (const int*)  d_in[0];
    const float* emb = (const float*)d_in[1];
    const float* cw  = (const float*)d_in[2];
    const float* cb  = (const float*)d_in[3];
    const float* W   = (const float*)d_in[4];
    const float* Bp  = (const float*)d_in[5];
    const float* acW = (const float*)d_in[6];
    const float* asW = (const float*)d_in[7];
    const float* opr = (const float*)d_in[8];
    const float* opi = (const float*)d_in[9];
    const float* Wr  = (const float*)d_in[10];
    const float* Wi  = (const float*)d_in[11];

    char* ws = (char*)d_ws;
    float2*       tab  = (float2*)(ws + OFF_TAB);
    float*        xr   = (float*) (ws + OFF_XR);
    float*        xi   = (float*) (ws + OFF_XI);
    float*        xc   = (float*) (ws + OFF_XC);
    float*        cosS = (float*) (ws + OFF_COS);
    float*        sinS = (float*) (ws + OFF_SIN);
    float4*       pk   = (float4*)(ws + OFF_PK);
    unsigned int* Ar   = (unsigned int*)(ws + OFF_AR);
    unsigned int* Ai   = (unsigned int*)(ws + OFF_AI);
    unsigned int* Br   = (unsigned int*)(ws + OFF_BR);
    unsigned int* Bi   = (unsigned int*)(ws + OFF_BI);

    build_tab_kernel<<<RES_SZ / 256, 256, 0, stream>>>(tab);
    coef_kernel<<<(L_SZ * N_SZ * D_SZ + 255) / 256, 256, 0, stream>>>(W, Bp, acW, asW, pk);
    prepB_kernel<<<(NT_V * KT_D * TILE_DW + 255) / 256, 256, 0, stream>>>(Wr, Wi, Br, Bi);
    embed_kernel<<<(BS_SZ * D_SZ + 255) / 256, 256, 0, stream>>>(ids, emb, xr, xi);

    for (int l = 0; l < L_SZ; ++l) {
        contract_kernel<<<BS_SZ, 128, 0, stream>>>(xr, xi, cw, cb, xc, l);
        attn_kernel<<<BS_SZ, 256, 0, stream>>>(xc, pk, tab, cosS, sinS, l);
        outproj_kernel<<<BS_SZ, 128, 0, stream>>>(cosS, sinS, opr, opi, xr, xi, l);
    }

    prepA_kernel<<<(MT_TOK * KT_D * TILE_DW + 255) / 256, 256, 0, stream>>>(xr, xi, Ar, Ai);
    vocab_kernel<<<dim3((NT_V + 15) / 16, MT_TOK / 2), 256, 0, stream>>>(Ar, Ai, Br, Bi,
                                                                         (float*)d_out);
}
